// BoTMHSA_3564822855739
// MI455X (gfx1250) — matrix operand order repaired and checked by reading, not compiled, not run on hardware
//
#include <hip/hip_runtime.h>

// BoT-MHSA for MI455X (gfx1250, wave32, WMMA + TDM double-buffered pipelines).
// ws layout (needs >= 96MB): Q bf16 [32][8][1024][64], K' bf16 (K + rel, same
// layout), Vt bf16 transposed [32][8][64][1024].

typedef __bf16 v16bf __attribute__((ext_vector_type(16)));
typedef __bf16 v8bf  __attribute__((ext_vector_type(8)));
typedef float  v8f   __attribute__((ext_vector_type(8)));
typedef unsigned int u32x4 __attribute__((ext_vector_type(4)));
typedef int          i32x8 __attribute__((ext_vector_type(8)));
typedef int          i32x4 __attribute__((ext_vector_type(4)));

#define N_BATCH 32
#define DIMC    512
#define NPIX    1024
#define N_HEADS 8
#define HD      64
#define SCALE   0.125f

// Padded LDS row widths (elements) chosen for bank-conflict-free column reads:
#define KS_W 80    // 64 bf16 + 8-dword TDM pad  -> 40-dword stride
#define VS_W 40    // 32 bf16 + 4-dword TDM pad  -> 20-dword stride (conflict-free)
#define XS_W 132   // 128 f32 + 4-dword TDM pad  -> bank = 4k + m, conflict-free
#define PL_W 40    // P relayout buffer, same 20-dword stride

__device__ inline v16bf ld8x2(const __bf16* p0, const __bf16* p1) {
  v8bf a = *(const v8bf*)p0;
  v8bf b = *(const v8bf*)p1;
  return __builtin_shufflevector(a, b, 0,1,2,3,4,5,6,7,8,9,10,11,12,13,14,15);
}

// ---------------------------------------------------------------------------
// TDM: 2D tile from global to LDS with optional LDS row padding.
// Builds D# group0/group1 per CDNA5 ISA; groups 2/3 zero.
//  dsize: 0=1B 1=2B 2=4B; dims/strides in elements.
//  pad_int code: interval = 2<<code dwords; pad_amt code: (code+1) dwords.
// ---------------------------------------------------------------------------
__device__ inline void tdm_load_2d(unsigned lds_off, const void* gaddr,
                                   unsigned dsize, unsigned dim0, unsigned dim1,
                                   unsigned tile0, unsigned tile1,
                                   unsigned stride0, unsigned pad_en,
                                   unsigned pad_int, unsigned pad_amt)
{
  const unsigned long long ga = (unsigned long long)(size_t)gaddr;
  u32x4 g0;
  g0[0] = 1u;                                           // count=1, user mode
  g0[1] = lds_off;                                      // lds_addr (bytes)
  g0[2] = (unsigned)(ga & 0xFFFFFFFFu);                 // global_addr[31:0]
  g0[3] = (unsigned)((ga >> 32) & 0x1FFFFFFu) | (2u << 30);  // [56:32] | type=2
  i32x8 g1;
  g1[0] = (int)((dsize << 16) | (pad_en << 20) | (pad_int << 22) | (pad_amt << 25));
  g1[1] = (int)((dim0 & 0xFFFFu) << 16);                // tensor_dim0 lo
  g1[2] = (int)(((dim0 >> 16) & 0xFFFFu) | ((dim1 & 0xFFFFu) << 16));
  g1[3] = (int)(((dim1 >> 16) & 0xFFFFu) | ((tile0 & 0xFFFFu) << 16));
  g1[4] = (int)(tile1 & 0xFFFFu);                       // tile_dim1, tile_dim2=0
  g1[5] = (int)stride0;                                 // tensor_dim0_stride lo
  g1[6] = 0;
  g1[7] = 0;
  const i32x4 z4 = {0, 0, 0, 0};
  const i32x8 z8 = {0, 0, 0, 0, 0, 0, 0, 0};
  __builtin_amdgcn_tensor_load_to_lds(g0, g1, z4, z4, z8, 0);
}

// ---------------------------------------------------------------------------
// Kernel 1: QKV projection (1x1 conv as GEMM) with bf16 WMMA.
//  A[m,k] = x[b, k, hw] staged through LDS by TDM (double-buffered);
//  B[k,o] = w_qkv[o, k] from global (3MB, L2-resident).
//  Epilogue: +bias; Q *= SCALE; K += rel(h,w); V stored transposed.
// ---------------------------------------------------------------------------
__global__ __launch_bounds__(256) void qkv_kernel(
    const float* __restrict__ x, const float* __restrict__ w,
    const float* __restrict__ bias, const float* __restrict__ hrel,
    const float* __restrict__ wrel, __bf16* __restrict__ Qb,
    __bf16* __restrict__ Kb, __bf16* __restrict__ Vt)
{
  __shared__ __align__(16) float Xs[2][32][XS_W];   // [buf][k][pixel], padded

  const int lane = threadIdx.x & 31;
  const int wave = threadIdx.x >> 5;
  const int half = lane >> 4;      // wave32: high 16 lanes
  const int l16  = lane & 15;
  const int m0   = blockIdx.x * 128;             // block pixel base (one image)
  const int o0   = blockIdx.y * 64;              // output-channel tile base
  const int bimg = m0 >> 10;
  const int mloc = wave * 16 + l16;              // lane's A row within tile

  const float* xtile = x + (size_t)bimg * DIMC * NPIX + (m0 & 1023); // + k*NPIX
  const unsigned ldsX[2] = {(unsigned)(size_t)(&Xs[0][0][0]),
                            (unsigned)(size_t)(&Xs[1][0][0])};

  if (wave == 0)
    tdm_load_2d(ldsX[0], xtile, 2, 128, 32, 128, 32, NPIX, 1, 6, 3);

  v8f acc[4] = {v8f{}, v8f{}, v8f{}, v8f{}};

  for (int kc = 0; kc < 16; ++kc) {              // 16 chunks of K=32
    if (wave == 0) __builtin_amdgcn_s_wait_tensorcnt(0);
    __syncthreads();                             // publish chunk kc; prev reads done
    if (wave == 0 && kc + 1 < 16)
      tdm_load_2d(ldsX[(kc + 1) & 1], xtile + (size_t)(kc + 1) * 32 * NPIX,
                  2, 128, 32, 128, 32, NPIX, 1, 6, 3);

    const float (*Xb)[XS_W] = Xs[kc & 1];
    const int k0  = kc * 32;
    const int klo = half * 16;
    // A fragment: lane<16 holds K {0..15}, lane>=16 holds {16..31}
    v16bf af;
    #pragma unroll
    for (int t = 0; t < 8; ++t) {
      af[t]     = (__bf16)Xb[klo + t][mloc];
      af[t + 8] = (__bf16)Xb[klo + 8 + t][mloc];
    }
    #pragma unroll
    for (int nb = 0; nb < 4; ++nb) {
      // B fragment: lane = column o, holds 16 contiguous k (half-selected)
      const float* wr = w + (size_t)(o0 + nb * 16 + l16) * DIMC + k0 + half * 16;
      v16bf bf_;
      #pragma unroll
      for (int t = 0; t < 16; ++t) bf_[t] = (__bf16)wr[t];
      acc[nb] = __builtin_amdgcn_wmma_f32_16x16x32_bf16(
          false, af, false, bf_, (short)0, acc[nb], false, false);
    }
  }

  // Epilogue: section is uniform per block (o0 is a multiple of 64)
  const int sec = o0 >> 9;              // 0=Q, 1=K, 2=V
  const int p   = (o0 & 511) >> 6;      // head
  const int m0w = m0 + wave * 16;

  #pragma unroll
  for (int nb = 0; nb < 4; ++nb) {
    const int d = nb * 16 + l16;
    const float bb = bias[o0 + nb * 16 + l16];
    #pragma unroll
    for (int r = 0; r < 8; ++r) {
      const int m  = m0w + r + half * 8; // C layout: lanes>=16 hold rows M+8
      const int bi = m >> 10;
      const int n  = m & 1023;
      const size_t bp = (size_t)bi * N_HEADS + p;
      float v = acc[nb][r] + bb;
      if (sec == 0) {
        Qb[(bp * NPIX + n) * HD + d] = (__bf16)(v * SCALE);
      } else if (sec == 1) {
        const int hh = n >> 5, ww = n & 31;
        v += hrel[(size_t)hh * DIMC + p * HD + d]
           + wrel[(size_t)ww * DIMC + p * HD + d];
        Kb[(bp * NPIX + n) * HD + d] = (__bf16)v;
      } else {
        Vt[(bp * HD + d) * NPIX + n] = (__bf16)v;
      }
    }
  }
}

// ---------------------------------------------------------------------------
// Kernel 2: flash attention per (batch, head). Block = 8 waves x 16 Q rows.
// 32-key chunks staged into double-buffered LDS by the TDM; DMA of chunk i+1
// overlaps WMMA+softmax of chunk i (one barrier per iteration).
// ---------------------------------------------------------------------------
__global__ __launch_bounds__(256) void attn_kernel(
    const __bf16* __restrict__ Qb, const __bf16* __restrict__ Kb,
    const __bf16* __restrict__ Vt, float* __restrict__ out)
{
  __shared__ __align__(16) __bf16 Ks[2][32][KS_W];    // [buf][key][d], padded
  __shared__ __align__(16) __bf16 Vs[2][64][VS_W];    // [buf][d][key], padded
  __shared__ __align__(16) __bf16 Pls[8][16][PL_W];   // per-wave P relayout

  const int lane = threadIdx.x & 31;
  const int wave = threadIdx.x >> 5;
  const int half = lane >> 4;
  const int l16  = lane & 15;
  const int q0   = blockIdx.x * 128 + wave * 16;  // this wave's Q rows
  const int p    = blockIdx.y;
  const int b    = blockIdx.z;
  const size_t bp = (size_t)b * N_HEADS + p;

  // Q fragments (A layout), row m = q0 + l16, pre-scaled by SCALE in kernel 1
  const __bf16* qrow = Qb + (bp * NPIX + q0 + l16) * HD;
  const int klo = half * 16;
  const v16bf qf0 = ld8x2(qrow + klo,      qrow + klo + 8);
  const v16bf qf1 = ld8x2(qrow + 32 + klo, qrow + 32 + klo + 8);

  v8f oc[4] = {v8f{}, v8f{}, v8f{}, v8f{}};
  float mrow[8], lrow[8];
  #pragma unroll
  for (int r = 0; r < 8; ++r) { mrow[r] = -1e30f; lrow[r] = 0.f; }

  const __bf16* Kbase = Kb + bp * NPIX * HD;
  const __bf16* Vbase = Vt + bp * HD * NPIX;
  const unsigned ldsK[2] = {(unsigned)(size_t)(&Ks[0][0][0]),
                            (unsigned)(size_t)(&Ks[1][0][0])};
  const unsigned ldsV[2] = {(unsigned)(size_t)(&Vs[0][0][0]),
                            (unsigned)(size_t)(&Vs[1][0][0])};

  if (wave == 0) {
    // K' chunk: 32 keys x 64 d (row 32 dwords + 8 pad -> 40-dword stride)
    tdm_load_2d(ldsK[0], Kbase, 1, 64, 32, 64, 32, 64, 1, 4, 7);
    // V chunk: 64 d x 32 keys (row 16 dwords + 4 pad -> 20-dword stride)
    tdm_load_2d(ldsV[0], Vbase, 1, 32, 64, 32, 64, NPIX, 1, 3, 3);
  }

  for (int it = 0; it < NPIX / 32; ++it) {
    if (wave == 0) __builtin_amdgcn_s_wait_tensorcnt(0);
    __syncthreads();                   // publish chunk it; prev chunk reads done
    if (wave == 0 && it + 1 < NPIX / 32) {
      const int kn = (it + 1) * 32;
      tdm_load_2d(ldsK[(it + 1) & 1], Kbase + (size_t)kn * HD,
                  1, 64, 32, 64, 32, 64, 1, 4, 7);
      tdm_load_2d(ldsV[(it + 1) & 1], Vbase + kn,
                  1, 32, 64, 32, 64, NPIX, 1, 3, 3);
    }
    const int bs = it & 1;

    // S = Q K'^T for two 16-key blocks; GEMM-K = d (64) in two wmma steps
    v8f s0 = {}, s1 = {};
    {
      const int doff = half * 16;
      v16bf b00 = ld8x2(&Ks[bs][l16][doff],           &Ks[bs][l16][doff + 8]);
      v16bf b01 = ld8x2(&Ks[bs][16 + l16][doff],      &Ks[bs][16 + l16][doff + 8]);
      s0 = __builtin_amdgcn_wmma_f32_16x16x32_bf16(false, qf0, false, b00, (short)0, s0, false, false);
      s1 = __builtin_amdgcn_wmma_f32_16x16x32_bf16(false, qf0, false, b01, (short)0, s1, false, false);
      v16bf b10 = ld8x2(&Ks[bs][l16][32 + doff],      &Ks[bs][l16][32 + doff + 8]);
      v16bf b11 = ld8x2(&Ks[bs][16 + l16][32 + doff], &Ks[bs][16 + l16][32 + doff + 8]);
      s0 = __builtin_amdgcn_wmma_f32_16x16x32_bf16(false, qf1, false, b10, (short)0, s0, false, false);
      s1 = __builtin_amdgcn_wmma_f32_16x16x32_bf16(false, qf1, false, b11, (short)0, s1, false, false);
    }

    // Online softmax: row reductions across a 16-lane half (wave32 shfl_xor)
    #pragma unroll
    for (int r = 0; r < 8; ++r) {
      float smax = fmaxf(s0[r], s1[r]);
      #pragma unroll
      for (int off = 8; off > 0; off >>= 1)
        smax = fmaxf(smax, __shfl_xor(smax, off, 32));
      const float mnew = fmaxf(mrow[r], smax);
      const float corr = __expf(mrow[r] - mnew);
      mrow[r] = mnew;
      const float p0 = __expf(s0[r] - mnew);
      const float p1 = __expf(s1[r] - mnew);
      float rs = p0 + p1;
      #pragma unroll
      for (int off = 8; off > 0; off >>= 1)
        rs += __shfl_xor(rs, off, 32);
      lrow[r] = lrow[r] * corr + rs;
      oc[0][r] *= corr; oc[1][r] *= corr; oc[2][r] *= corr; oc[3][r] *= corr;
      const int m = r + half * 8;                  // C-layout row
      Pls[wave][m][l16]      = (__bf16)p0;
      Pls[wave][m][16 + l16] = (__bf16)p1;
    }
    asm volatile("s_wait_dscnt 0" ::: "memory");   // P stores -> cross-lane reads

    // O += P @ V  (GEMM-K = 32 keys), B columns = d, from transposed Vs rows
    const v16bf pa = ld8x2(&Pls[wave][l16][klo], &Pls[wave][l16][klo + 8]);
    #pragma unroll
    for (int f = 0; f < 4; ++f) {
      v16bf bv = ld8x2(&Vs[bs][f * 16 + l16][half * 16],
                       &Vs[bs][f * 16 + l16][half * 16 + 8]);
      oc[f] = __builtin_amdgcn_wmma_f32_16x16x32_bf16(false, pa, false, bv, (short)0, oc[f], false, false);
    }
  }

  // Normalize and write out[b, p*64+d, h, w]  (address = d*1024 + n)
  float* obase = out + ((size_t)b * DIMC + (size_t)p * HD) * NPIX;
  #pragma unroll
  for (int r = 0; r < 8; ++r) {
    const float inv = 1.0f / lrow[r];
    const int n = q0 + r + half * 8;
    #pragma unroll
    for (int f = 0; f < 4; ++f)
      obase[(size_t)(f * 16 + l16) * NPIX + n] = oc[f][r] * inv;
  }
}

// ---------------------------------------------------------------------------
extern "C" void kernel_launch(void* const* d_in, const int* in_sizes, int n_in,
                              void* d_out, int out_size, void* d_ws, size_t ws_size,
                              hipStream_t stream) {
  const float* x     = (const float*)d_in[0];
  const float* w_qkv = (const float*)d_in[1];
  const float* b_qkv = (const float*)d_in[2];
  const float* h_rel = (const float*)d_in[3];
  const float* w_rel = (const float*)d_in[4];
  float* out = (float*)d_out;

  const size_t tsz = (size_t)N_BATCH * N_HEADS * NPIX * HD;  // 16.7M elems
  __bf16* Qb = (__bf16*)d_ws;
  __bf16* Kb = Qb + tsz;
  __bf16* Vt = Kb + tsz;

  qkv_kernel<<<dim3(256, 24), 256, 0, stream>>>(x, w_qkv, b_qkv, h_rel, w_rel,
                                                Qb, Kb, Vt);
  attn_kernel<<<dim3(8, N_HEADS, N_BATCH), 256, 0, stream>>>(Qb, Kb, Vt, out);
}
